// AqtDotGeneral_64828236366182
// MI455X (gfx1250) — 8-bit matrix operand order repaired and checked by reading, compiled, not run on hardware
//
#include <hip/hip_runtime.h>
#include <stdint.h>

typedef int v8i __attribute__((ext_vector_type(8)));

#define M_DIM 8192
#define K_DIM 4096
#define N_DIM 4096
#define KFRAG 64
#define KB    (K_DIM / KFRAG)
#define QMAXF 127.0f
#define QDIVF 127.5f

#define SL_OFF   0u
#define SR_OFF   32768u
#define PART_OFF 49152u
#define QA_OFF   262144u
#define QB_OFF   (262144u + 33554432u)

__global__ void __launch_bounds__(256) k_row_absmax(const float* __restrict__ lhs,
                                                    float* __restrict__ sl) {
    __shared__ float red[256];
    const int m = blockIdx.x;
    const float* row = lhs + (size_t)m * K_DIM;
    float mx = 0.0f;
    for (int k = threadIdx.x; k < K_DIM; k += 256) mx = fmaxf(mx, fabsf(row[k]));
    red[threadIdx.x] = mx;
    __syncthreads();
    for (int s = 128; s > 0; s >>= 1) {
        if (threadIdx.x < s) red[threadIdx.x] = fmaxf(red[threadIdx.x], red[threadIdx.x + s]);
        __syncthreads();
    }
    if (threadIdx.x == 0) {
        float s = red[0] / QDIVF;
        sl[m] = (s == 0.0f) ? 1.0f : s;
    }
}

__global__ void __launch_bounds__(256) k_col_partial(const float* __restrict__ rhs,
                                                     float* __restrict__ part) {
    const int c = blockIdx.x * 256 + threadIdx.x;
    const int r0 = blockIdx.y * (K_DIM / 8);
    float mx = 0.0f;
    for (int r = r0; r < r0 + (K_DIM / 8); ++r)
        mx = fmaxf(mx, fabsf(rhs[(size_t)r * N_DIM + c]));
    part[(size_t)blockIdx.y * N_DIM + c] = mx;
}

__global__ void __launch_bounds__(256) k_col_final(const float* __restrict__ part,
                                                   float* __restrict__ sr) {
    const int c = blockIdx.x * 256 + threadIdx.x;
    float mx = 0.0f;
    for (int i = 0; i < 8; ++i) mx = fmaxf(mx, part[(size_t)i * N_DIM + c]);
    float s = mx / QDIVF;
    sr[c] = (s == 0.0f) ? 1.0f : s;
}

__global__ void __launch_bounds__(256) k_quant_lhs(const float* __restrict__ lhs,
                                                   const float* __restrict__ sl,
                                                   int8_t* __restrict__ qa) {
    const size_t idx = (size_t)blockIdx.x * 256 + threadIdx.x;
    const int m = (int)(idx >> 12);
    const int k = (int)(idx & 4095);
    float q = rintf(lhs[idx] / sl[m]);
    q = fminf(fmaxf(q, -QMAXF), QMAXF);
    const int kl = k & 63;
    const int h  = (kl >> 4) & 1;
    const int v  = ((kl >> 5) & 1) * 4 + ((kl >> 2) & 3);
    const int lane = (m & 15) + 16 * h;
    const size_t frag = (size_t)(m >> 4) * KB + (size_t)(k >> 6);
    qa[frag * 1024 + (size_t)lane * 32 + v * 4 + (kl & 3)] = (int8_t)(int)q;
}

__global__ void __launch_bounds__(256) k_quant_rhs(const float* __restrict__ rhs,
                                                   const float* __restrict__ sr,
                                                   int8_t* __restrict__ qb) {
    const size_t idx = (size_t)blockIdx.x * 256 + threadIdx.x;
    const int k = (int)(idx >> 12);
    const int n = (int)(idx & 4095);
    float q = rintf(rhs[idx] / sr[n]);
    q = fminf(fmaxf(q, -QMAXF), QMAXF);
    const int kl = k & 63;
    const int h  = (kl >> 4) & 1;
    const int v  = ((kl >> 5) & 1) * 4 + ((kl >> 2) & 3);
    const int lane = (n & 15) + 16 * h;
    const size_t frag = (size_t)(n >> 4) * KB + (size_t)(k >> 6);
    qb[frag * 1024 + (size_t)lane * 32 + v * 4 + (kl & 3)] = (int8_t)(int)q;
}

__global__ void __launch_bounds__(256) k_gemm(const int8_t* __restrict__ qa,
                                              const int8_t* __restrict__ qb,
                                              const float* __restrict__ sl,
                                              const float* __restrict__ sr,
                                              float* __restrict__ out) {
    const int tid  = threadIdx.x;
    const int lane = tid & 31;
    const int wave = tid >> 5;
    const int wm   = wave >> 2;
    const int wn   = wave & 3;
    const int m0   = blockIdx.y * 128 + wm * 64;
    const int n0   = blockIdx.x * 256 + wn * 64;
    const int mt0  = m0 >> 4;
    const int nt0  = n0 >> 4;

    const v8i* A = (const v8i*)qa;
    const v8i* B = (const v8i*)qb;
    size_t abase[4], bbase[4];
#pragma unroll
    for (int i = 0; i < 4; ++i) {
        abase[i] = (size_t)(mt0 + i) * KB * 32 + lane;
        bbase[i] = (size_t)(nt0 + i) * KB * 32 + lane;
    }

    v8i acc[4][4];
#pragma unroll
    for (int i = 0; i < 4; ++i)
#pragma unroll
        for (int j = 0; j < 4; ++j) acc[i][j] = (v8i){0, 0, 0, 0, 0, 0, 0, 0};

#pragma unroll 1
    for (int kb = 0; kb < KB; ++kb) {
        v8i fb[4];
#pragma unroll
        for (int j = 0; j < 4; ++j) fb[j] = B[bbase[j] + (size_t)kb * 32];
#pragma unroll
        for (int i = 0; i < 4; ++i) {
            const v8i fa = A[abase[i] + (size_t)kb * 32];
#pragma unroll
            for (int j = 0; j < 4; ++j)
                acc[i][j] = __builtin_amdgcn_wmma_i32_16x16x64_iu8(true, fa, true, fb[j],
                                                                   acc[i][j], false, false);
        }
    }

    const int h  = lane >> 4;
    const int nl = lane & 15;
    float srj[4];
#pragma unroll
    for (int j = 0; j < 4; ++j) srj[j] = sr[n0 + j * 16 + nl];
#pragma unroll
    for (int i = 0; i < 4; ++i) {
#pragma unroll
        for (int r = 0; r < 8; ++r) {
            const int row = m0 + i * 16 + r + 8 * h;
            const float slr = sl[row];
#pragma unroll
            for (int j = 0; j < 4; ++j)
                out[(size_t)row * N_DIM + n0 + j * 16 + nl] = (float)acc[i][j][r] * slr * srj[j];
        }
    }
}

extern "C" void kernel_launch(void* const* d_in, const int* in_sizes, int n_in,
                              void* d_out, int out_size, void* d_ws, size_t ws_size,
                              hipStream_t stream) {
    const float* lhs = (const float*)d_in[0];
    const float* rhs = (const float*)d_in[1];
    float* out = (float*)d_out;

    char* ws = (char*)d_ws;
    float*  sl   = (float*)(ws + SL_OFF);
    float*  sr   = (float*)(ws + SR_OFF);
    float*  part = (float*)(ws + PART_OFF);
    int8_t* qa   = (int8_t*)(ws + QA_OFF);
    int8_t* qb   = (int8_t*)(ws + QB_OFF);

    k_row_absmax<<<M_DIM, 256, 0, stream>>>(lhs, sl);
    k_col_partial<<<dim3(N_DIM / 256, 8), 256, 0, stream>>>(rhs, part);
    k_col_final<<<N_DIM / 256, 256, 0, stream>>>(part, sr);

    k_quant_lhs<<<(M_DIM * (size_t)K_DIM) / 256, 256, 0, stream>>>(lhs, sl, qa);
    k_quant_rhs<<<((size_t)K_DIM * N_DIM) / 256, 256, 0, stream>>>(rhs, sr, qb);

    k_gemm<<<dim3(N_DIM / 256, M_DIM / 128), 256, 0, stream>>>(qa, qb, sl, sr, out);
}
